// FastSpeech_86182813761977
// MI455X (gfx1250) — matrix operand order repaired and checked by reading, not compiled, not run on hardware
//
#include <hip/hip_runtime.h>
#include <cstdint>

typedef __attribute__((ext_vector_type(16))) _Float16 v16h;
typedef __attribute__((ext_vector_type(8)))  _Float16 v8h;
typedef __attribute__((ext_vector_type(8)))  float    v8f;
typedef __attribute__((ext_vector_type(4)))  float    v4f;

// ---- model constants ----
constexpr int NB    = 4;      // batch
constexpr int NT    = 128;    // token seq len
constexpr int ND    = 512;    // model dim
constexpr int NH    = 8;      // heads
constexpr int NDH   = 64;     // head dim
constexpr int NMM   = 1408;   // MAXMEL
constexpr int NMELS = 80;
constexpr int NDPF  = 256;
constexpr float ASCALE = 0.125f;   // 1/sqrt(64)

enum { GF_RELU = 1, GF_F16 = 2 };

__device__ __forceinline__ v8f v8f_zero() {
  v8f z = {0.f,0.f,0.f,0.f,0.f,0.f,0.f,0.f};
  return z;
}

// One contiguous 32B fragment read: 2x ds_load_b128. p must be 16B aligned.
__device__ __forceinline__ v16h ldfrag(const _Float16* p) {
  const v8h* q = (const v8h*)p;
  v8h lo = q[0], hi = q[1];
  return __builtin_shufflevector(lo, hi, 0,1,2,3,4,5,6,7,8,9,10,11,12,13,14,15);
}

// =====================================================================
// Generic WMMA GEMM / K-tap conv1d (SAME):
//   out[b,s,:] = act( sum_tap A[b, s+tap-1, :] @ W[tap] + bias )
// A fp32 [B*S, Cin]; W fp32 [ntaps, Cin, Cout].
// Tile 256x64; 8 waves x (32 rows x 64 cols): 2 A-frags reuse 4 B-frags
// -> 8 WMMA per K-step per wave. aF rows in contiguous K order
// (k = 16*half + s); bTt transposed so B fragments are contiguous.
// =====================================================================
__global__ __launch_bounds__(256) void gemm_conv_kernel(
    const float* __restrict__ A, const float* __restrict__ W,
    const float* __restrict__ bias, float* __restrict__ Of,
    _Float16* __restrict__ Oh,
    int S, int Cin, int Cout, int ntaps, int flags)
{
  __shared__ __align__(16) _Float16 aF[256][40];   // 80B stride
  __shared__ __align__(16) _Float16 bTt[64][40];   // [col][k]

  const int tid  = threadIdx.x;
  const int lane = tid & 31;
  const int wave = tid >> 5;          // 0..7
  const int half = lane >> 4;         // 0/1
  const int l16  = lane & 15;
  const int mBlk = blockIdx.x * 256;
  const int nBlk = blockIdx.y * 64;

  v8f acc[2][4];
#pragma unroll
  for (int m = 0; m < 2; ++m)
#pragma unroll
    for (int i = 0; i < 4; ++i) acc[m][i] = v8f_zero();

  for (int tap = 0; tap < ntaps; ++tap) {
    const int soff = (ntaps == 3) ? (tap - 1) : 0;
    const float* Wt = W + (size_t)tap * Cin * Cout;
    for (int k0 = 0; k0 < Cin; k0 += 32) {
      // --- stage A tile: one full 32-K row per thread, chunks in K order ---
      {
        const int row = tid;
        const int r  = mBlk + row;
        const int b  = r / S;
        const int s2 = (r % S) + soff;
        const bool ok = (s2 >= 0) && (s2 < S);
        v4f t[8];
#pragma unroll
        for (int q = 0; q < 8; ++q) t[q] = (v4f){0,0,0,0};
        if (ok) {
          const v4f* src = (const v4f*)(A + ((size_t)b * S + s2) * Cin + k0);
#pragma unroll
          for (int q = 0; q < 8; ++q) t[q] = src[q];
        }
        v8h h[4];
#pragma unroll
        for (int c = 0; c < 4; ++c)
#pragma unroll
          for (int j = 0; j < 4; ++j) {
            h[c][j]     = (_Float16)t[2 * c][j];
            h[c][4 + j] = (_Float16)t[2 * c + 1][j];
          }
        v8h* dst = (v8h*)&aF[row][0];
        dst[0] = h[0];   // K0-7   -> chunk 0
        dst[1] = h[1];   // K8-15  -> chunk 1
        dst[2] = h[2];   // K16-23 -> chunk 2
        dst[3] = h[3];   // K24-31 -> chunk 3
      }
      // --- stage B tile transposed (32x64 -> bTt[col][k]) ---
      {
        const int row   = tid >> 3;          // k within tile
        const int cbase = (tid & 7) * 8;     // col group
        const float* src = Wt + (size_t)(k0 + row) * Cout + nBlk + cbase;
#pragma unroll
        for (int j = 0; j < 8; ++j) {
          const int c = nBlk + cbase + j;
          bTt[cbase + j][row] = (c < Cout) ? (_Float16)src[j] : (_Float16)0.0f;
        }
      }
      __syncthreads();
      const v16h af0 = ldfrag(&aF[wave * 32 + l16][half * 16]);
      const v16h af1 = ldfrag(&aF[wave * 32 + 16 + l16][half * 16]);
#pragma unroll
      for (int nf = 0; nf < 4; ++nf) {
        const v16h bf = ldfrag(&bTt[nf * 16 + l16][half * 16]);
        acc[0][nf] = __builtin_amdgcn_wmma_f32_16x16x32_f16(
            false, af0, false, bf, (short)0, acc[0][nf], false, false);
        acc[1][nf] = __builtin_amdgcn_wmma_f32_16x16x32_f16(
            false, af1, false, bf, (short)0, acc[1][nf], false, false);
      }
      __syncthreads();
    }
  }
  // --- epilogue: bias, relu, store (fp32 or f16) ---
#pragma unroll
  for (int m = 0; m < 2; ++m) {
    const int rowb = mBlk + wave * 32 + m * 16 + half * 8;
#pragma unroll
    for (int nf = 0; nf < 4; ++nf) {
      const int col = nBlk + nf * 16 + l16;
      if (col < Cout) {
        const float bv = bias ? bias[col] : 0.0f;
#pragma unroll
        for (int r = 0; r < 8; ++r) {
          float v = acc[m][nf][r] + bv;
          if (flags & GF_RELU) v = fmaxf(v, 0.0f);
          const size_t idx = (size_t)(rowb + r) * Cout + col;
          if (flags & GF_F16) Oh[idx] = (_Float16)v;
          else                Of[idx] = v;
        }
      }
    }
  }
}

// =====================================================================
// Flash attention: grid (S/64, B*H), 4 waves; Q/K/V f16 [B*S, 512].
// qT: rows in contiguous dh order. kT: K-row-major (B-contiguous for
// Q.K^T). vTt: transposed (B-contiguous for P.V). pT: written in
// plain key order (pos == col), read contiguously.
// =====================================================================
__global__ __launch_bounds__(128) void attn_kernel(
    const _Float16* __restrict__ Q, const _Float16* __restrict__ K,
    const _Float16* __restrict__ V, const int* __restrict__ mask,
    float* __restrict__ O, int S)
{
  __shared__ __align__(16) _Float16 qT[64][72];       // 144B stride
  __shared__ __align__(16) _Float16 kT[64][72];
  __shared__ __align__(16) _Float16 vTt[64][72];      // [dh col][key]
  __shared__ __align__(16) _Float16 pT[4][16][72];

  const int bh = blockIdx.y;
  const int b  = bh / NH, h = bh % NH;
  const int qBase = blockIdx.x * 64;
  const int tid  = threadIdx.x;
  const int lane = tid & 31;
  const int wave = tid >> 5;
  const int half = lane >> 4;
  const int l16  = lane & 15;

  // load Q tile, contiguous chunks within each 32-half group
  {
    const int row = tid >> 1, cb = (tid & 1) * 32;
    const v8h* src = (const v8h*)(Q + ((size_t)(b * S + qBase + row)) * ND + h * NDH + cb);
    v8h* dst = (v8h*)&qT[row][0];
    const int g4 = (cb >> 5) * 4;
    dst[g4 + 0] = src[0];
    dst[g4 + 1] = src[1];
    dst[g4 + 2] = src[2];
    dst[g4 + 3] = src[3];
  }

  v8f o[4];
#pragma unroll
  for (int i = 0; i < 4; ++i) o[i] = v8f_zero();
  float m_i[8], l_i[8];
#pragma unroll
  for (int r = 0; r < 8; ++r) { m_i[r] = -3.0e38f; l_i[r] = 0.0f; }

  for (int kt = 0; kt < S; kt += 64) {
    __syncthreads();   // previous iter done with kT/vTt
    {
      const int row = tid >> 1, cb = (tid & 1) * 32;
      const v8h* sk = (const v8h*)(K + ((size_t)(b * S + kt + row)) * ND + h * NDH + cb);
      const v8h* sv = (const v8h*)(V + ((size_t)(b * S + kt + row)) * ND + h * NDH + cb);
      v8h* kd = (v8h*)&kT[row][cb];
#pragma unroll
      for (int q = 0; q < 4; ++q) kd[q] = sk[q];
#pragma unroll
      for (int q = 0; q < 4; ++q) {
        const v8h t = sv[q];
#pragma unroll
        for (int j = 0; j < 8; ++j) vTt[cb + q * 8 + j][row] = t[j];
      }
    }
    __syncthreads();

    // ---- scores = Q (16xDH) x K^T (DHx64), DH=64 in two k-steps ----
    v8f s[4];
#pragma unroll
    for (int i = 0; i < 4; ++i) s[i] = v8f_zero();
#pragma unroll
    for (int ks = 0; ks < 64; ks += 32) {
      const v16h af = ldfrag(&qT[wave * 16 + l16][ks + half * 16]);
#pragma unroll
      for (int nf = 0; nf < 4; ++nf) {
        const v16h bf = ldfrag(&kT[nf * 16 + l16][ks + half * 16]);
        s[nf] = __builtin_amdgcn_wmma_f32_16x16x32_f16(
            false, af, false, bf, (short)0, s[nf], false, false);
      }
    }

    // ---- scale + key-pad mask + online softmax ----
    float rmax[8];
#pragma unroll
    for (int r = 0; r < 8; ++r) rmax[r] = -3.0e38f;
#pragma unroll
    for (int nf = 0; nf < 4; ++nf) {
      const int key = kt + nf * 16 + l16;
      const float mb = (mask[b * S + key] != 0) ? -1.0e9f : 0.0f;
#pragma unroll
      for (int r = 0; r < 8; ++r) {
        const float v = s[nf][r] * ASCALE + mb;
        s[nf][r] = v;
        rmax[r] = fmaxf(rmax[r], v);
      }
    }
#pragma unroll
    for (int off = 1; off < 16; off <<= 1)
#pragma unroll
      for (int r = 0; r < 8; ++r)
        rmax[r] = fmaxf(rmax[r], __shfl_xor(rmax[r], off, 32));

    float alpha[8], rsum[8];
#pragma unroll
    for (int r = 0; r < 8; ++r) {
      const float mn = fmaxf(m_i[r], rmax[r]);
      alpha[r] = __expf(m_i[r] - mn);
      m_i[r] = mn;
      rsum[r] = 0.0f;
    }
#pragma unroll
    for (int nf = 0; nf < 4; ++nf)
#pragma unroll
      for (int r = 0; r < 8; ++r) {
        const float p = __expf(s[nf][r] - m_i[r]);
        s[nf][r] = p;
        rsum[r] += p;
      }
#pragma unroll
    for (int off = 1; off < 16; off <<= 1)
#pragma unroll
      for (int r = 0; r < 8; ++r)
        rsum[r] += __shfl_xor(rsum[r], off, 32);
#pragma unroll
    for (int r = 0; r < 8; ++r) l_i[r] = l_i[r] * alpha[r] + rsum[r];

    // P stored in plain key order (pos == col) (wave-private region)
#pragma unroll
    for (int nf = 0; nf < 4; ++nf) {
      const int col = nf * 16 + l16;
      const int c   = (col >> 3) & 3;
      const int pcc = c;   // {0,1,2,3}
      const int pos = (col & 7) + 8 * pcc + 32 * (col >> 5);
#pragma unroll
      for (int r = 0; r < 8; ++r)
        pT[wave][half * 8 + r][pos] = (_Float16)s[nf][r];
    }
#pragma unroll
    for (int nf = 0; nf < 4; ++nf)
#pragma unroll
      for (int r = 0; r < 8; ++r)
        o[nf][r] *= alpha[r];

    // ---- O += P (16x64) x V (64x64) ---- (same-wave LDS RAW: DS in-order)
#pragma unroll
    for (int ks = 0; ks < 64; ks += 32) {
      const v16h af = ldfrag(&pT[wave][l16][ks + half * 16]);
#pragma unroll
      for (int nf = 0; nf < 4; ++nf) {
        const v16h bf = ldfrag(&vTt[nf * 16 + l16][ks + half * 16]);
        o[nf] = __builtin_amdgcn_wmma_f32_16x16x32_f16(
            false, af, false, bf, (short)0, o[nf], false, false);
      }
    }
  }

  const int rowb = qBase + wave * 16 + half * 8;
#pragma unroll
  for (int nf = 0; nf < 4; ++nf) {
    const int col = h * NDH + nf * 16 + l16;
#pragma unroll
    for (int r = 0; r < 8; ++r)
      O[(size_t)(b * S + rowb + r) * ND + col] = o[nf][r] / l_i[r];
  }
}

// =====================================================================
// LayerNorm (optional residual add, optional pad-row zeroing). 1 wave/row.
// =====================================================================
template <int NCOL>
__global__ __launch_bounds__(32) void ln_kernel(
    const float* __restrict__ x, const float* __restrict__ y,
    const float* __restrict__ g, const float* __restrict__ bt,
    float* __restrict__ out, const int* __restrict__ mask)
{
  const int row = blockIdx.x;
  const int lane = threadIdx.x;
  constexpr int PER = NCOL / 32;
  float vals[PER];
  const float* xr = x + (size_t)row * NCOL;
  const float* yr = y ? y + (size_t)row * NCOL : nullptr;
  float sum = 0.f;
#pragma unroll
  for (int j = 0; j < PER; ++j) {
    float v = xr[lane + 32 * j];
    if (yr) v += yr[lane + 32 * j];
    vals[j] = v; sum += v;
  }
#pragma unroll
  for (int off = 1; off < 32; off <<= 1) sum += __shfl_xor(sum, off, 32);
  const float mean = sum * (1.0f / NCOL);
  float vs = 0.f;
#pragma unroll
  for (int j = 0; j < PER; ++j) { const float d = vals[j] - mean; vs += d * d; }
#pragma unroll
  for (int off = 1; off < 32; off <<= 1) vs += __shfl_xor(vs, off, 32);
  const float inv = rsqrtf(vs * (1.0f / NCOL) + 1e-5f);
  const float mz = (mask && mask[row] != 0) ? 0.0f : 1.0f;
#pragma unroll
  for (int j = 0; j < PER; ++j) {
    const int c = lane + 32 * j;
    out[(size_t)row * NCOL + c] = ((vals[j] - mean) * inv * g[c] + bt[c]) * mz;
  }
}

// ---- duration predictor head: d = h @ lw + lb, masked ----
__global__ __launch_bounds__(32) void dp_final_kernel(
    const float* __restrict__ h, const float* __restrict__ lw,
    const float* __restrict__ lb, const int* __restrict__ mask,
    float* __restrict__ dur)
{
  const int row = blockIdx.x;
  const int lane = threadIdx.x;
  float s = 0.f;
  for (int c = lane; c < NDPF; c += 32) s += h[(size_t)row * NDPF + c] * lw[c];
#pragma unroll
  for (int off = 1; off < 32; off <<= 1) s += __shfl_xor(s, off, 32);
  if (lane == 0) dur[row] = (mask[row] != 0) ? 0.0f : (s + lb[0]);
}

// ---- per-batch start/end prefix sums over target durations ----
__global__ void cumsum_kernel(const int* __restrict__ dur,
                              int* __restrict__ st, int* __restrict__ en)
{
  const int b = blockIdx.x;
  if (threadIdx.x == 0) {
    int acc = 0;
    for (int t = 0; t < NT; ++t) {
      st[b * NT + t] = acc;
      acc += dur[b * NT + t];
      en[b * NT + t] = acc;
    }
  }
}

// ---- length regulation: y[b,m,:]=enc[b,t,:] for the unique active t; align ----
__global__ __launch_bounds__(128) void regulate_kernel(
    const float* __restrict__ enc, const int* __restrict__ st,
    const int* __restrict__ en, float* __restrict__ y,
    float* __restrict__ align)
{
  const int m = blockIdx.x;
  const int b = blockIdx.y;
  const int tid = threadIdx.x;
  __shared__ int sel;
  if (tid == 0) sel = -1;
  __syncthreads();
  if (tid < NT) {
    const bool in = (m >= st[b * NT + tid]) && (m < en[b * NT + tid]);
    align[((size_t)b * NMM + m) * NT + tid] = in ? 1.0f : 0.0f;
    if (in) sel = tid;
  }
  __syncthreads();
  const int t = sel;
  for (int c = tid; c < ND; c += 128)
    y[((size_t)b * NMM + m) * ND + c] =
        (t >= 0) ? enc[((size_t)b * NT + t) * ND + c] : 0.0f;
}

// ---- embedding gather ----
__global__ __launch_bounds__(128) void embed_kernel(
    const int* __restrict__ tok, const float* __restrict__ emb,
    float* __restrict__ x)
{
  const int r = blockIdx.x;
  const int t = tok[r];
  for (int c = threadIdx.x; c < ND; c += 128)
    x[(size_t)r * ND + c] = emb[(size_t)t * ND + c];
}

// =====================================================================
extern "C" void kernel_launch(void* const* d_in, const int* in_sizes, int n_in,
                              void* d_out, int out_size, void* d_ws, size_t ws_size,
                              hipStream_t stream) {
  if (n_in < 113) return;
  const int Renc = NB * NT;     // 512
  const int Rdec = NB * NMM;    // 5632

  const int*   tokens = (const int*)d_in[0];
  const int*   tdur   = (const int*)d_in[1];
  const int*   emask  = (const int*)d_in[2];
  const int*   dmask  = (const int*)d_in[3];
  const float* emb    = (const float*)d_in[4];

  // workspace carve (~53 MB)
  char* w = (char*)d_ws;
  float* bufX = (float*)w;  w += (size_t)Rdec * ND * 4;
  float* bufA = (float*)w;  w += (size_t)Rdec * ND * 4;
  float* bufB = (float*)w;  w += (size_t)Rdec * ND * 4;
  _Float16* qh = (_Float16*)w; w += (size_t)Rdec * ND * 2;
  _Float16* kh = (_Float16*)w; w += (size_t)Rdec * ND * 2;
  _Float16* vh = (_Float16*)w; w += (size_t)Rdec * ND * 2;
  float* dp1 = (float*)w;   w += (size_t)Renc * NDPF * 4;
  float* dp2 = (float*)w;   w += (size_t)Renc * NDPF * 4;
  int* starts = (int*)w;    w += NB * NT * 4;
  int* ends   = (int*)w;    w += NB * NT * 4;

  float* out_mels  = (float*)d_out;                       // [Rdec, 80]
  float* out_dur   = out_mels + (size_t)Rdec * NMELS;     // [512]
  float* out_align = out_dur + Renc;                      // [Rdec, 128]

  auto gemm = [&](const float* A, const float* W, const float* bias,
                  float* Of, _Float16* Oh, int S, int Cin, int Cout,
                  int ntaps, int flags) {
    dim3 g((NB * S) / 256, (Cout + 63) / 64);
    gemm_conv_kernel<<<g, 256, 0, stream>>>(A, W, bias, Of, Oh,
                                            S, Cin, Cout, ntaps, flags);
  };

  // FFT block: X in/out, Ta/Tb scratch. Param base index pb into d_in.
  auto fft = [&](float* X, float* Ta, float* Tb, int S, const int* mask, int pb) {
    auto P = [&](int k) { return (const float*)d_in[pb + k]; };
    const int R = NB * S;
    gemm(X, P(0), P(1), nullptr, qh, S, ND, ND, 1, GF_F16);      // Q
    gemm(X, P(2), P(3), nullptr, kh, S, ND, ND, 1, GF_F16);      // K
    gemm(X, P(4), P(5), nullptr, vh, S, ND, ND, 1, GF_F16);      // V
    attn_kernel<<<dim3(S / 64, NB * NH), 128, 0, stream>>>(qh, kh, vh, mask, Ta, S);
    gemm(Ta, P(6), P(7), Tb, nullptr, S, ND, ND, 1, 0);          // out proj
    ln_kernel<ND><<<R, 32, 0, stream>>>(X, Tb, P(8), P(9), X, nullptr);
    gemm(X,  P(10), P(11), Ta, nullptr, S, ND, ND, 3, GF_RELU);  // conv1+relu
    gemm(Ta, P(12), P(13), Tb, nullptr, S, ND, ND, 3, 0);        // conv2
    ln_kernel<ND><<<R, 32, 0, stream>>>(X, Tb, P(14), P(15), X, mask);
  };

  // ---- encoder ----
  embed_kernel<<<Renc, 128, 0, stream>>>(tokens, emb, bufX);
  for (int l = 0; l < 3; ++l) fft(bufX, bufA, bufB, NT, emask, 5 + 16 * l);

  // ---- duration predictor (pb = 101) ----
  gemm(bufX, (const float*)d_in[101], (const float*)d_in[102], dp1, nullptr,
       NT, ND, NDPF, 3, GF_RELU);
  ln_kernel<NDPF><<<Renc, 32, 0, stream>>>(dp1, nullptr,
      (const float*)d_in[103], (const float*)d_in[104], dp1, nullptr);
  gemm(dp1, (const float*)d_in[105], (const float*)d_in[106], dp2, nullptr,
       NT, NDPF, NDPF, 3, GF_RELU);
  ln_kernel<NDPF><<<Renc, 32, 0, stream>>>(dp2, nullptr,
      (const float*)d_in[107], (const float*)d_in[108], dp2, nullptr);
  dp_final_kernel<<<Renc, 32, 0, stream>>>(dp2,
      (const float*)d_in[109], (const float*)d_in[110], emask, out_dur);

  // ---- length regulation (uses target durations) ----
  cumsum_kernel<<<NB, 32, 0, stream>>>(tdur, starts, ends);
  regulate_kernel<<<dim3(NMM, NB), 128, 0, stream>>>(bufX, starts, ends,
                                                     bufA, out_align);

  // ---- decoder (enc x in bufX is dead; reuse as scratch) ----
  for (int l = 0; l < 3; ++l) fft(bufA, bufX, bufB, NMM, dmask, 53 + 16 * l);

  // ---- mel projection ----
  gemm(bufA, (const float*)d_in[111], (const float*)d_in[112], out_mels, nullptr,
       NMM, ND, NMELS, 1, 0);

  (void)in_sizes; (void)out_size; (void)ws_size;
}
